// CrossAttention_57698590654516
// MI455X (gfx1250) — matrix operand order repaired and checked by reading, not compiled, not run on hardware
//
#include <hip/hip_runtime.h>

// ---------------------------------------------------------------------------
// CrossAttention on MI455X (gfx1250, wave32, WMMA bf16 -> f32 accum)
// B=4, N=4096, L=1024, C=1024, H=16, D=64
// Round 2: 64x32 register-blocked projection GEMMs, 32-query flash tiles,
//          global_prefetch hints on the K/V stream.
// ---------------------------------------------------------------------------

#define DEVINL __device__ __forceinline__

typedef __bf16 bf16_t;
typedef __attribute__((ext_vector_type(8)))  __bf16 v8bf;
typedef __attribute__((ext_vector_type(16))) __bf16 v16bf;
typedef __attribute__((ext_vector_type(8)))  float  v8f;

constexpr int kB = 4, kN = 4096, kL = 1024, kC = 1024, kH = 16, kD = 64;
constexpr float kScale = 0.125f; // 1/sqrt(64)

DEVINL v16bf cat16(v8bf lo, v8bf hi) {
  v16bf r;
#pragma unroll
  for (int i = 0; i < 8; ++i) { r[i] = lo[i]; r[i + 8] = hi[i]; }
  return r;
}
DEVINL v8bf ld8(const bf16_t* p) { return *(const v8bf*)p; }
DEVINL bf16_t f2bf(float f) { return (bf16_t)f; }

DEVINL v8f wmma_bf16(v16bf a, v16bf b, v8f c) {
  return __builtin_amdgcn_wmma_f32_16x16x32_bf16(false, a, false, b,
                                                 (short)0, c, false, false);
}

// ---------------------------------------------------------------------------
// elementwise f32 -> bf16 cast
// ---------------------------------------------------------------------------
__global__ void cast_f32_bf16(const float* __restrict__ src,
                              bf16_t* __restrict__ dst, int n) {
  int i = blockIdx.x * blockDim.x + threadIdx.x;
  int stride = gridDim.x * blockDim.x;
  for (; i < n; i += stride) dst[i] = f2bf(src[i]);
}

// ---------------------------------------------------------------------------
// 64x32 output block of  act(MxK) @ W(NxK)^T : 4 A-frags x 2 B-frags,
// 8 WMMAs per 32-wide k-step (0.75 fragment fetches per WMMA vs 2 unblocked).
// Fragment layouts per ISA 7.12.2:
//   A (16x32 bf16): lane L -> row (L&15), k chunks at 8*(L>>4) and +16.
//   B (32x16 bf16): lane L -> col (L&15), k chunks at 8*(L>>4) and +16 (same as A).
//   C (16x16 f32):  lane L -> col (L&15), rows r + 8*(L>>4).
// ---------------------------------------------------------------------------
DEVINL void mma_block_64x32(const bf16_t* __restrict__ A, int lda,
                            const bf16_t* __restrict__ W, int ldw,
                            int K, int lane, v8f acc[4][2]) {
  const int asel = (lane >> 4) * 8;
  const int bsel = (lane >> 4) * 8;
  const bf16_t* arow0 = A + (size_t)(lane & 15) * lda + asel;
  const bf16_t* wrow0 = W + (size_t)(lane & 15) * ldw + bsel;
  for (int k0 = 0; k0 < K; k0 += 32) {
    v16bf bfrag[2];
#pragma unroll
    for (int j = 0; j < 2; ++j) {
      const bf16_t* wr = wrow0 + (size_t)j * 16 * ldw + k0;
      bfrag[j] = cat16(ld8(wr), ld8(wr + 16));
    }
#pragma unroll
    for (int i = 0; i < 4; ++i) {
      const bf16_t* ar = arow0 + (size_t)i * 16 * lda + k0;
      v16bf a = cat16(ld8(ar), ld8(ar + 16));
#pragma unroll
      for (int j = 0; j < 2; ++j) acc[i][j] = wmma_bf16(a, bfrag[j], acc[i][j]);
    }
  }
}

// ---------------------------------------------------------------------------
// Q projection: q[m][n] = query[m][:] . Wq[n][:],  M=B*L, N=K=C. bf16 out.
// ---------------------------------------------------------------------------
__global__ void qproj_kernel(const bf16_t* __restrict__ act,
                             const bf16_t* __restrict__ Wq,
                             bf16_t* __restrict__ out) {
  int lane = threadIdx.x & 31;
  int ntile = blockIdx.x;  // 32-col blocks
  int mtile = blockIdx.y;  // 64-row blocks
  v8f acc[4][2];
#pragma unroll
  for (int i = 0; i < 4; ++i)
#pragma unroll
    for (int j = 0; j < 2; ++j)
#pragma unroll
      for (int r = 0; r < 8; ++r) acc[i][j][r] = 0.0f;
  mma_block_64x32(act + (size_t)mtile * 64 * kC, kC,
                  Wq + (size_t)ntile * 32 * kC, kC, kC, lane, acc);
#pragma unroll
  for (int i = 0; i < 4; ++i)
#pragma unroll
    for (int j = 0; j < 2; ++j) {
      int col = ntile * 32 + j * 16 + (lane & 15);
      int mbase = mtile * 64 + i * 16 + (lane >> 4) * 8;
#pragma unroll
      for (int r = 0; r < 8; ++r)
        out[(size_t)(mbase + r) * kC + col] = f2bf(acc[i][j][r]);
    }
}

// ---------------------------------------------------------------------------
// KV projection: cols [0,C) -> K stored [B,H,N,D]; cols [C,2C) -> V stored
// transposed [B,H,D,N]. Region is uniform per 32-col block (1024 % 32 == 0).
// ---------------------------------------------------------------------------
__global__ void kvproj_kernel(const bf16_t* __restrict__ x,
                              const bf16_t* __restrict__ Wkv,
                              bf16_t* __restrict__ kdst,
                              bf16_t* __restrict__ vTdst) {
  int lane = threadIdx.x & 31;
  int ntile = blockIdx.x;  // 64 blocks over 2C cols
  int mtile = blockIdx.y;  // 256 blocks over B*N rows
  v8f acc[4][2];
#pragma unroll
  for (int i = 0; i < 4; ++i)
#pragma unroll
    for (int j = 0; j < 2; ++j)
#pragma unroll
      for (int r = 0; r < 8; ++r) acc[i][j][r] = 0.0f;
  mma_block_64x32(x + (size_t)mtile * 64 * kC, kC,
                  Wkv + (size_t)ntile * 32 * kC, kC, kC, lane, acc);
#pragma unroll
  for (int i = 0; i < 4; ++i)
#pragma unroll
    for (int j = 0; j < 2; ++j) {
      int col = ntile * 32 + j * 16 + (lane & 15);        // 0..2047
      int mbase = mtile * 64 + i * 16 + (lane >> 4) * 8;  // batch-aligned
      int b = mbase >> 12;                                // /N
      int nrow = mbase & (kN - 1);
      if (col < kC) {                                     // K part: [B,H,N,D]
        int h = col >> 6, d = col & 63;
        bf16_t* kp = kdst + ((size_t)(b * kH + h) * kN) * kD + d;
#pragma unroll
        for (int r = 0; r < 8; ++r) kp[(size_t)(nrow + r) * kD] = f2bf(acc[i][j][r]);
      } else {                                            // V part: [B,H,D,N]
        int c = col - kC;
        int h = c >> 6, d = c & 63;
        v8bf pk;
#pragma unroll
        for (int r = 0; r < 8; ++r) pk[r] = f2bf(acc[i][j][r]);
        *(v8bf*)(vTdst + ((size_t)(b * kH + h) * kD + d) * kN + nrow) = pk;
      }
    }
}

// ---------------------------------------------------------------------------
// Flash attention, one wave per (b, h, 32-query tile). Two query subtiles
// share every K/V fragment -> 16 WMMAs per 8 fragment loads per 32-key step,
// and total K/V L2 traffic is halved vs 16-query tiles.
// S^T = K_tile @ Q^T keeps each query's scores in one lane (plus xor-16
// partner), so softmax is in-lane + one shuffle. O^T += V^T @ P^T.
// ---------------------------------------------------------------------------
__global__ void attn_kernel(const bf16_t* __restrict__ qmat,
                            const bf16_t* __restrict__ kmat,
                            const bf16_t* __restrict__ vTmat,
                            bf16_t* __restrict__ attn_out) {
  int lane = threadIdx.x & 31;
  int wave = threadIdx.x >> 5;
  int w = blockIdx.x * (blockDim.x >> 5) + wave;  // 0..B*H*(L/32)-1
  int mt = w & 31;   // 32-query tile
  int bh = w >> 5;
  int b = bh >> 4, h = bh & 15;

  const bf16_t* qp = qmat + ((size_t)b * kL + mt * 32) * kC + h * kD;
  const bf16_t* kp = kmat + (size_t)bh * kN * kD;
  const bf16_t* vp = vTmat + (size_t)bh * kD * kN;

  const int asel = (lane >> 4) * 8;
  const int n0 = lane & 15;
  const bool hi = lane >= 16;

  // Q^T B-operands for the two query subtiles, loaded once.
  v16bf qb[2][2];
#pragma unroll
  for (int qt = 0; qt < 2; ++qt) {
    const bf16_t* qrow = qp + (size_t)(qt * 16 + n0) * kC + (lane >> 4) * 8;
    qb[qt][0] = cat16(ld8(qrow), ld8(qrow + 16));       // d 0..31
    qb[qt][1] = cat16(ld8(qrow + 32), ld8(qrow + 48));  // d 32..63
  }

  v8f oacc[2][4];
#pragma unroll
  for (int qt = 0; qt < 2; ++qt)
#pragma unroll
    for (int t = 0; t < 4; ++t)
#pragma unroll
      for (int i = 0; i < 8; ++i) oacc[qt][t][i] = 0.0f;
  float run_m[2] = {-1e30f, -1e30f}, run_l[2] = {0.0f, 0.0f};

  for (int nb = 0; nb < kN; nb += 32) {
    // prefetch the K/V stream ~2 steps ahead (global_prefetch_b8, speculative)
    int pf = (nb + 64 <= kN - 32) ? nb + 64 : nb;
    __builtin_prefetch(kp + (size_t)(pf + n0) * kD + asel, 0, 3);
    __builtin_prefetch(vp + (size_t)(lane * 2) * kN + pf, 0, 3);

    // --- scores: two 16x16 S^T subtiles per query subtile, K = d = 64 ---
    v8f st[2][2];
#pragma unroll
    for (int qt = 0; qt < 2; ++qt)
#pragma unroll
      for (int s = 0; s < 2; ++s)
#pragma unroll
        for (int i = 0; i < 8; ++i) st[qt][s][i] = 0.0f;
#pragma unroll
    for (int sub = 0; sub < 2; ++sub) {
      const bf16_t* krow = kp + (size_t)(nb + sub * 16 + n0) * kD + asel;
#pragma unroll
      for (int ds = 0; ds < 2; ++ds) {
        v16bf a = cat16(ld8(krow + ds * 32), ld8(krow + ds * 32 + 16));
#pragma unroll
        for (int qt = 0; qt < 2; ++qt)
          st[qt][sub] = wmma_bf16(a, qb[qt][ds], st[qt][sub]);
      }
    }

    // --- online softmax + P^T B-operand build, per query subtile ---
    v16bf pb[2];
#pragma unroll
    for (int qt = 0; qt < 2; ++qt) {
      float p0[8], p1[8];
      float lm = -1e30f;
#pragma unroll
      for (int i = 0; i < 8; ++i) {
        p0[i] = st[qt][0][i] * kScale;
        p1[i] = st[qt][1][i] * kScale;
        lm = fmaxf(lm, fmaxf(p0[i], p1[i]));
      }
      lm = fmaxf(lm, __shfl_xor(lm, 16, 32));
      float new_m = fmaxf(run_m[qt], lm);
      float alpha = __expf(run_m[qt] - new_m);
      float ls = 0.0f;
#pragma unroll
      for (int i = 0; i < 8; ++i) {
        p0[i] = __expf(p0[i] - new_m);
        p1[i] = __expf(p1[i] - new_m);
        ls += p0[i] + p1[i];
      }
      ls += __shfl_xor(ls, 16, 32);
      run_l[qt] = run_l[qt] * alpha + ls;
      run_m[qt] = new_m;
#pragma unroll
      for (int t = 0; t < 4; ++t)
#pragma unroll
        for (int i = 0; i < 8; ++i) oacc[qt][t][i] *= alpha;
      // C-layout -> B-layout rearrange: own 8 values + 8 from lane^16.
#pragma unroll
      for (int i = 0; i < 8; ++i) {
        float t0x = __shfl_xor(p0[i], 16, 32);
        float t1x = __shfl_xor(p1[i], 16, 32);
        pb[qt][i]     = f2bf(hi ? t1x : p0[i]);
        pb[qt][i + 8] = f2bf(hi ? p1[i] : t0x);
      }
    }

    // --- O^T += V^T @ P^T: 4 d-tiles, V-frag shared by both query subtiles ---
#pragma unroll
    for (int t = 0; t < 4; ++t) {
      const bf16_t* vrow = vp + (size_t)(t * 16 + n0) * kN + nb + (lane >> 4) * 16;
      v16bf av = cat16(ld8(vrow), ld8(vrow + 8));
#pragma unroll
      for (int qt = 0; qt < 2; ++qt)
        oacc[qt][t] = wmma_bf16(av, pb[qt], oacc[qt][t]);
    }
  }

  // --- finalize: divide by l, pack 8 consecutive d per b128 store ---
#pragma unroll
  for (int qt = 0; qt < 2; ++qt) {
    float inv_l = 1.0f / run_l[qt];
    bf16_t* op = attn_out +
                 ((size_t)b * kL + mt * 32 + qt * 16 + n0) * kC +
                 h * kD + (lane >> 4) * 8;
#pragma unroll
    for (int t = 0; t < 4; ++t) {
      v8bf pk;
#pragma unroll
      for (int r = 0; r < 8; ++r) pk[r] = f2bf(oacc[qt][t][r] * inv_l);
      *(v8bf*)(op + t * 16) = pk;
    }
  }
}

// ---------------------------------------------------------------------------
// Output projection + bias, f32 result to d_out.
// ---------------------------------------------------------------------------
__global__ void outproj_kernel(const bf16_t* __restrict__ act,
                               const bf16_t* __restrict__ Wp,
                               const float* __restrict__ bias,
                               float* __restrict__ out) {
  int lane = threadIdx.x & 31;
  int ntile = blockIdx.x;
  int mtile = blockIdx.y;
  v8f acc[4][2];
#pragma unroll
  for (int i = 0; i < 4; ++i)
#pragma unroll
    for (int j = 0; j < 2; ++j)
#pragma unroll
      for (int r = 0; r < 8; ++r) acc[i][j][r] = 0.0f;
  mma_block_64x32(act + (size_t)mtile * 64 * kC, kC,
                  Wp + (size_t)ntile * 32 * kC, kC, kC, lane, acc);
#pragma unroll
  for (int i = 0; i < 4; ++i)
#pragma unroll
    for (int j = 0; j < 2; ++j) {
      int col = ntile * 32 + j * 16 + (lane & 15);
      float bv = bias[col];
      int mbase = mtile * 64 + i * 16 + (lane >> 4) * 8;
#pragma unroll
      for (int r = 0; r < 8; ++r)
        out[(size_t)(mbase + r) * kC + col] = acc[i][j][r] + bv;
    }
}

// ---------------------------------------------------------------------------
extern "C" void kernel_launch(void* const* d_in, const int* in_sizes, int n_in,
                              void* d_out, int out_size, void* d_ws,
                              size_t ws_size, hipStream_t stream) {
  const float* x     = (const float*)d_in[0];
  const float* query = (const float*)d_in[1];
  const float* Wq    = (const float*)d_in[2];
  const float* Wkv   = (const float*)d_in[3];
  const float* Wproj = (const float*)d_in[4];
  const float* bproj = (const float*)d_in[5];

  char* ws = (char*)d_ws;
  size_t off = 0;
  auto alloc = [&](size_t bytes) { char* p = ws + off; off += bytes; return p; };
  bf16_t* query_bf = (bf16_t*)alloc((size_t)kB * kL * kC * 2);
  bf16_t* x_bf     = (bf16_t*)alloc((size_t)kB * kN * kC * 2);
  bf16_t* Wq_bf    = (bf16_t*)alloc((size_t)kC * kC * 2);
  bf16_t* Wkv_bf   = (bf16_t*)alloc((size_t)2 * kC * kC * 2);
  bf16_t* Wp_bf    = (bf16_t*)alloc((size_t)kC * kC * 2);
  bf16_t* q_bf     = (bf16_t*)alloc((size_t)kB * kL * kC * 2);
  bf16_t* k_bf     = (bf16_t*)alloc((size_t)kB * kH * kN * kD * 2);
  bf16_t* vT_bf    = (bf16_t*)alloc((size_t)kB * kH * kD * kN * 2);
  bf16_t* attn_bf  = (bf16_t*)alloc((size_t)kB * kL * kC * 2);
  (void)ws_size; // total = 128 MB

  // 1) casts to bf16
  cast_f32_bf16<<<4096, 256, 0, stream>>>(query, query_bf, kB * kL * kC);
  cast_f32_bf16<<<4096, 256, 0, stream>>>(x, x_bf, kB * kN * kC);
  cast_f32_bf16<<<1024, 256, 0, stream>>>(Wq, Wq_bf, kC * kC);
  cast_f32_bf16<<<2048, 256, 0, stream>>>(Wkv, Wkv_bf, 2 * kC * kC);
  cast_f32_bf16<<<1024, 256, 0, stream>>>(Wproj, Wp_bf, kC * kC);

  // 2) Q projection: [B*L, C] x [C, C]^T, 64x32 blocks
  qproj_kernel<<<dim3(kC / 32, kB * kL / 64), 32, 0, stream>>>(query_bf, Wq_bf, q_bf);

  // 3) KV projection: [B*N, C] x [2C, C]^T -> K [B,H,N,D], V^T [B,H,D,N]
  kvproj_kernel<<<dim3(2 * kC / 32, kB * kN / 64), 32, 0, stream>>>(x_bf, Wkv_bf, k_bf, vT_bf);

  // 4) flash attention: one wave per 32-query tile; 8 waves/block
  attn_kernel<<<dim3(kB * kH * (kL / 32) / 8), 256, 0, stream>>>(q_bf, k_bf, vT_bf, attn_bf);

  // 5) output projection + bias -> f32
  outproj_kernel<<<dim3(kC / 32, kB * kL / 64), 32, 0, stream>>>(attn_bf, Wp_bf, bproj, (float*)d_out);
}
